// MinimalNetwork_56607668962065
// MI455X (gfx1250) — compile-verified, matrix operand order checked by reading, not run on hardware
//
#include <hip/hip_runtime.h>
#include <math.h>

#define N_NODES 10000
#define N_EDGES 100000
#define NP      128     // padded hidden width
#define KP      128     // padded K for layers 1..3
#define LDR     321     // LDS row stride for R tile (max 320 cols)

typedef __attribute__((ext_vector_type(16))) __bf16 v16bf;
typedef __attribute__((ext_vector_type(8)))  __bf16 v8bf;
typedef __attribute__((ext_vector_type(8)))  float  v8f;

// ---------------------------------------------------------------------------
// helpers
// ---------------------------------------------------------------------------
__device__ __forceinline__ v16bf ld16(const __bf16* p0, const __bf16* p1) {
  v8bf a = *(const v8bf*)p0;
  v8bf b = *(const v8bf*)p1;
  v16bf r;
#pragma unroll
  for (int i = 0; i < 8; i++) { r[i] = a[i]; r[i + 8] = b[i]; }
  return r;
}

__device__ __forceinline__ v8f wmma_bf16(v16bf a, v16bf b, v8f c) {
  // D = A(16x32 bf16) * B(32x16 bf16) + C(f32)
  return __builtin_amdgcn_wmma_f32_16x16x32_bf16(false, a, false, b,
                                                 (short)0, c, false, false);
}

// ---------------------------------------------------------------------------
// Clebsch-Gordan table generation (runs once per launch, deterministic)
// ---------------------------------------------------------------------------
__device__ double dfact(int n) { double r = 1.0; for (int i = 2; i <= n; i++) r *= (double)i; return r; }

__device__ double w3j(int j1, int j2, int j3, int m1, int m2, int m3) {
  if (m1 + m2 + m3 != 0) return 0.0;
  int dj = j1 - j2; if (dj < 0) dj = -dj;
  if (j3 < dj || j3 > j1 + j2) return 0.0;
  if (abs(m1) > j1 || abs(m2) > j2 || abs(m3) > j3) return 0.0;
  double pre = sqrt(dfact(j1 + j2 - j3) * dfact(j1 - j2 + j3) * dfact(-j1 + j2 + j3)
                    / dfact(j1 + j2 + j3 + 1)
                    * dfact(j1 + m1) * dfact(j1 - m1) * dfact(j2 + m2) * dfact(j2 - m2)
                    * dfact(j3 + m3) * dfact(j3 - m3));
  int kmin = 0;
  if (j2 - j3 - m1 > kmin) kmin = j2 - j3 - m1;
  if (j1 - j3 + m2 > kmin) kmin = j1 - j3 + m2;
  int kmax = j1 + j2 - j3;
  if (j1 - m1 < kmax) kmax = j1 - m1;
  if (j2 + m2 < kmax) kmax = j2 + m2;
  double s = 0.0;
  for (int k = kmin; k <= kmax; k++) {
    double d = dfact(k) * dfact(j1 + j2 - j3 - k) * dfact(j1 - m1 - k)
             * dfact(j2 + m2 - k) * dfact(j3 - j2 + m1 + k) * dfact(j3 - j1 - m2 + k);
    s += ((k & 1) ? -1.0 : 1.0) / d;
  }
  double sign = ((j1 - j2 - m3) & 1) ? -1.0 : 1.0;
  return sign * pre * s;
}

__device__ void uc2r(int l, int a, int b, double* re, double* im) {
  int mr = a - l, mc = b - l;
  *re = 0.0; *im = 0.0;
  const double is2 = 0.7071067811865476;
  if (mr == 0) { if (mc == 0) *re = 1.0; return; }
  if (mr > 0) {
    if (mc == mr)       *re = (mr & 1) ? -is2 : is2;
    else if (mc == -mr) *re = is2;
  } else {
    int m = -mr;
    if (mc == -m)      *im = is2;            //  i/sqrt2
    else if (mc == m)  *im = (m & 1) ? is2 : -is2; // -i*(-1)^m/sqrt2
  }
}

__global__ void k_cg_init(float* __restrict__ cg) {
  const int lo_[19] = {0,0,0,1,1,1,1,1,1,1,2,2,2,2,2,2,2,2,2};
  const int li_[19] = {0,1,2,0,1,1,1,2,2,2,0,1,1,1,2,2,2,2,2};
  const int l_ [19] = {0,1,2,1,0,1,2,1,2,3,2,1,2,3,0,1,2,3,4};
  const int off_[19]= {0,1,10,35,44,53,80,125,170,245,350,375,420,495,600,625,700,825,1000};
  int gid = blockIdx.x * blockDim.x + threadIdx.x;
  if (gid >= 1225) return;
  int t = 18;
  for (int s = 0; s < 19; s++) { if (gid < off_[s]) { t = s - 1; break; } }
  int l1 = lo_[t], l2 = li_[t], l3 = l_[t];
  int P = 2 * l1 + 1, Q = 2 * l2 + 1, Rr = 2 * l3 + 1;
  int local = gid - off_[t];
  int a = local / (Q * Rr);
  int b = (local / Rr) % Q;
  int c = local % Rr;
  double accR = 0.0, accI = 0.0;
  for (int m1 = 0; m1 < P; m1++)
    for (int m2 = 0; m2 < Q; m2++)
      for (int m3 = 0; m3 < Rr; m3++) {
        double w = w3j(l1, l2, l3, m1 - l1, m2 - l2, m3 - l3);
        if (w == 0.0) continue;
        double r1, i1, r2, i2, r3, i3;
        uc2r(l1, a, m1, &r1, &i1);
        uc2r(l2, b, m2, &r2, &i2);
        uc2r(l3, c, m3, &r3, &i3);
        double rr = r1 * r2 - i1 * i2, ri = r1 * i2 + i1 * r2;
        double fr = rr * r3 - ri * i3, fi = rr * i3 + ri * r3;
        accR += fr * w; accI += fi * w;
      }
  cg[gid] = (float)(accR + accI);
}

// ---------------------------------------------------------------------------
// Weight convert: f32 [K][N] -> bf16 transposed [Np][Kp], prescaled by 1/sqrt(K)
// ---------------------------------------------------------------------------
__global__ void k_cvt(const float* __restrict__ W, __bf16* __restrict__ Bt,
                      int K, int N, int Kp, int Np, float scale) {
  int idx = blockIdx.x * blockDim.x + threadIdx.x;
  if (idx >= Kp * Np) return;
  int n = idx / Kp, k = idx % Kp;
  float v = (k < K && n < N) ? W[k * N + n] * scale : 0.0f;
  Bt[idx] = (__bf16)v;
}

// ---------------------------------------------------------------------------
// Per-edge: gaussian radial basis (bf16, padded to 32) + real sph-harm Y (25 f32)
// ---------------------------------------------------------------------------
__global__ void k_edge_prep(const float* __restrict__ dists, const float* __restrict__ rel,
                            __bf16* __restrict__ basis, float* __restrict__ Yb) {
  int e = blockIdx.x * blockDim.x + threadIdx.x;
  if (e >= N_EDGES) return;
  float d = dists[e];
#pragma unroll
  for (int k = 0; k < 32; k++) {
    float v = 0.0f;
    if (k < 10) {
      float c = 0.7f + 2.5f * (float)k / 9.0f;
      float t = (d - c) * 4.5f;                 // 1/sigma, sigma = 0.8*2.5/9
      v = expf(-t * t) * (1.0f / 1.423085244900308f);
    }
    basis[(size_t)e * 32 + k] = (__bf16)v;
  }
  float x = rel[e * 3 + 0], y = rel[e * 3 + 1], z = rel[e * 3 + 2];
  float r = sqrtf(x * x + y * y + z * z);
  float inv = 1.0f / fmaxf(r, 1e-9f);
  x *= inv; y *= inv; z *= inv;
  float A[5], B[5];
  A[0] = 1.0f; B[0] = 0.0f;
#pragma unroll
  for (int m = 1; m <= 4; m++) { A[m] = x * A[m-1] - y * B[m-1]; B[m] = x * B[m-1] + y * A[m-1]; }
  float P[5][5] = {};
  P[0][0] = 1.0f;
#pragma unroll
  for (int m = 0; m <= 4; m++) {
    if (m > 0)     P[m][m]     = -(2.0f * m - 1.0f) * P[m-1][m-1];
    if (m + 1 <= 4) P[m+1][m]  =  (2.0f * m + 1.0f) * z * P[m][m];
#pragma unroll
    for (int l = m + 2; l <= 4; l++)
      P[l][m] = ((2.0f * l - 1.0f) * z * P[l-1][m] - (float)(l + m - 1) * P[l-2][m]) / (float)(l - m);
  }
  const float F_[9] = {1.f,1.f,2.f,6.f,24.f,120.f,720.f,5040.f,40320.f};
  const float FOURPI = 12.566370614359172f;
  int idx = 0;
#pragma unroll
  for (int l = 0; l <= 4; l++) {
#pragma unroll
    for (int m = -l; m <= l; m++) {
      int am = (m < 0) ? -m : m;
      float Kn = sqrtf((float)(2 * l + 1) / FOURPI * F_[l - am] / F_[l + am]);
      float val;
      if (m == 0)      val = Kn * P[l][0];
      else if (m > 0)  val = 1.4142135623730951f * Kn * P[l][m]  * A[m];
      else             val = 1.4142135623730951f * Kn * P[l][am] * B[am];
      Yb[(size_t)e * 25 + idx] = val;
      idx++;
    }
  }
}

__global__ void k_zero(float* __restrict__ out, int n) {
  int i = blockIdx.x * blockDim.x + threadIdx.x;
  if (i < n) out[i] = 0.0f;
}

// ---------------------------------------------------------------------------
// GEMM + silu: C[E][128] = silu(A[E][Kp] @ Bt^T), all bf16 in, bf16 out, f32 acc
// block = 256 threads = 8 waves; wave w owns 16x16 output tile (cols w*16..)
// ---------------------------------------------------------------------------
__global__ void __launch_bounds__(256) k_gemm_silu(const __bf16* __restrict__ A,
                                                   const __bf16* __restrict__ Bt,
                                                   __bf16* __restrict__ C, int Kp) {
  int tid = threadIdx.x;
  int lane = tid & 31, wid = tid >> 5;
  int mr = lane & 15, hs = lane >> 4;
  int e0 = blockIdx.x * 16;
  const __bf16* arow = A  + (size_t)(e0 + mr) * Kp;
  const __bf16* brow = Bt + (size_t)(wid * 16 + mr) * Kp;
  v8f acc = {};
  int nk = Kp >> 5;
  for (int kk = 0; kk < nk; kk++) {
    int k0 = kk * 32 + hs * 8;
    v16bf a = ld16(arow + k0, arow + k0 + 16);
    v16bf b = ld16(brow + k0, brow + k0 + 16);
    acc = wmma_bf16(a, b, acc);
  }
#pragma unroll
  for (int rr = 0; rr < 8; rr++) {
    int m = rr + 8 * hs;
    float v = acc[rr];
    v = v / (1.0f + expf(-v));                  // silu
    C[(size_t)(e0 + m) * NP + wid * 16 + mr] = (__bf16)v;
  }
}

// ---------------------------------------------------------------------------
// Fused: R = h3 @ W3t (WMMA, per (lo,li) block through LDS) + CG tensor product
// + atomic segment-sum scatter.  block = 128 threads (4 waves), 16 edges/block.
// ---------------------------------------------------------------------------
__global__ void __launch_bounds__(128) k_tp(const __bf16* __restrict__ h3,
                                            const __bf16* __restrict__ w3t,
                                            const float* __restrict__ cg,
                                            const float* __restrict__ Yb,
                                            const float* __restrict__ x,
                                            const int* __restrict__ ei,
                                            float* __restrict__ out) {
  __shared__ float R_s[16 * LDR];
  __shared__ float F_s[16 * 72];
  __shared__ float Y_s[16 * 25];
  __shared__ float cg_s[1225];
  __shared__ int   src_s[16];
  __shared__ int   dst_s[16];

  const int TSTART[3][3] = {{0,1,2},{3,4,7},{10,11,14}};
  const int TRIOFF[19]   = {0,1,10,35,44,53,80,125,170,245,350,375,420,495,600,625,700,825,1000};
  const int RBASE[3][3]  = {{0,64,128},{192,256,448},{640,704,896}};
  const int FB[3]        = {0,8,32};

  int tid = threadIdx.x;
  int e0  = blockIdx.x * 16;
  if (tid < 16) { src_s[tid] = ei[e0 + tid]; dst_s[tid] = ei[N_EDGES + e0 + tid]; }
  __syncthreads();
  for (int idx = tid; idx < 16 * 72; idx += 128) {
    int e = idx / 72, c = idx % 72;
    F_s[idx] = x[(size_t)src_s[e] * 72 + c];
  }
  for (int idx = tid; idx < 16 * 25; idx += 128) Y_s[idx] = Yb[(size_t)e0 * 25 + idx];
  for (int idx = tid; idx < 1225;   idx += 128) cg_s[idx] = cg[idx];

  int lane = tid & 31, wid = tid >> 5, hs = lane >> 4, mr = lane & 15;
  const __bf16* arow = h3 + (size_t)(e0 + mr) * KP;
  v16bf afr[4];
#pragma unroll
  for (int kk = 0; kk < 4; kk++) {
    int k0 = kk * 32 + hs * 8;
    afr[kk] = ld16(arow + k0, arow + k0 + 16);
  }

  float vacc[9];
#pragma unroll
  for (int it = 0; it < 9; it++) vacc[it] = 0.0f;
  int e_me = tid & 15;       // my edge
  int cb   = tid >> 4;       // my comp base; comps = cb + 8*it

  for (int i = 0; i < 3; i++) {
    for (int j = 0; j < 3; j++) {
      int nlm = 2 * ((i < j) ? i : j) + 1;
      __syncthreads();       // also covers setup before first phase
      // ---- WMMA phase: R block (i,j) -> LDS, ncols = 64*nlm ----
      int ntiles = 4 * nlm;
      for (int ct = wid; ct < ntiles; ct += 4) {
        int ngl = RBASE[i][j] + ct * 16 + mr;
        const __bf16* brow = w3t + (size_t)ngl * KP;
        v8f acc = {};
#pragma unroll
        for (int kk = 0; kk < 4; kk++) {
          int k0 = kk * 32 + hs * 8;
          v16bf b = ld16(brow + k0, brow + k0 + 16);
          acc = wmma_bf16(afr[kk], b, acc);
        }
#pragma unroll
        for (int rr = 0; rr < 8; rr++)
          R_s[(rr + 8 * hs) * LDR + ct * 16 + mr] = acc[rr];
      }
      __syncthreads();
      // ---- tensor-product partial for out-block i, in-block j ----
      int lmin = (i > j) ? (i - j) : (j - i);
      int tj = 2 * j + 1;
      for (int it = 0; it < 9; it++) {
        int comp = cb + 8 * it;
        int bi, u, p;
        if (comp < 8)        { bi = 0; u = comp;        p = 0; }
        else if (comp < 32)  { int c2 = comp - 8;  bi = 1; u = c2 / 3; p = c2 % 3; }
        else                 { int c2 = comp - 32; bi = 2; u = c2 / 5; p = c2 % 5; }
        if (bi != i) continue;
        float stot = 0.0f;
        for (int k = 0; k < nlm; k++) {
          int l = lmin + k, tl = 2 * l + 1;
          const float* Cc = cg_s + TRIOFF[TSTART[i][j] + k];
          const float* Ye = Y_s + e_me * 25 + l * l;
          float cy[5];
          for (int q = 0; q < tj; q++) {
            float s = 0.0f;
            const float* Cr = Cc + (p * tj + q) * tl;
            for (int rr = 0; rr < tl; rr++) s += Cr[rr] * Ye[rr];
            cy[q] = s;
          }
          const float* Fe = F_s + e_me * 72 + FB[j];
          const float* Re = R_s + e_me * LDR + k;
          float s2 = 0.0f;
          for (int v = 0; v < 8; v++) {
            float fv = 0.0f;
            for (int q = 0; q < tj; q++) fv += Fe[v * tj + q] * cy[q];
            s2 += Re[(u * 8 + v) * nlm] * fv;
          }
          stot += s2;
        }
        vacc[it] += stot;
      }
    }
  }
  // ---- scatter (segment_sum over dst) ----
  const float NSE[3] = {24.0f, 56.0f, 72.0f};
  int dst = dst_s[e_me];
#pragma unroll
  for (int it = 0; it < 9; it++) {
    int comp = cb + 8 * it;
    int bi = (comp < 8) ? 0 : ((comp < 32) ? 1 : 2);
    float nrm = 3.5449077018110318f * sqrtf(2.0f * bi + 1.0f) / sqrtf(NSE[bi]);
    atomicAdd(out + (size_t)dst * 72 + comp, vacc[it] * nrm);
  }
}

// ---------------------------------------------------------------------------
// launch
// ---------------------------------------------------------------------------
extern "C" void kernel_launch(void* const* d_in, const int* in_sizes, int n_in,
                              void* d_out, int out_size, void* d_ws, size_t ws_size,
                              hipStream_t stream) {
  (void)in_sizes; (void)n_in; (void)out_size; (void)ws_size;
  const int*   ei = (const int*)d_in[0];
  const float* x  = (const float*)d_in[1];
  const float* dd = (const float*)d_in[2];
  const float* rv = (const float*)d_in[3];
  const float* W0 = (const float*)d_in[4];
  const float* W1 = (const float*)d_in[5];
  const float* W2 = (const float*)d_in[6];
  const float* W3 = (const float*)d_in[7];
  float* out = (float*)d_out;
  char* ws = (char*)d_ws;

  float*  cg    = (float*)(ws + 0);           // 1225 f32
  __bf16* wb0   = (__bf16*)(ws + 5120);       // 128x32 bf16
  __bf16* wb1   = (__bf16*)(ws + 13312);      // 128x128
  __bf16* wb2   = (__bf16*)(ws + 46080);      // 128x128
  __bf16* wb3   = (__bf16*)(ws + 78848);      // 1216x128
  __bf16* basis = (__bf16*)(ws + 390144);     // E x 32
  __bf16* h1    = (__bf16*)(ws + 6790144);    // E x 128
  __bf16* h2    = (__bf16*)(ws + 32390144);   // E x 128
  __bf16* h3    = (__bf16*)(ws + 57990144);   // E x 128
  float*  Yb    = (float*)(ws + 83590144);    // E x 25

  k_cg_init<<<5, 256, 0, stream>>>(cg);
  k_cvt<<<(32 * 128 + 255) / 256, 256, 0, stream>>>(W0, wb0, 10, 100, 32, 128, 0.31622776601683794f);
  k_cvt<<<(128 * 128 + 255) / 256, 256, 0, stream>>>(W1, wb1, 100, 100, 128, 128, 0.1f);
  k_cvt<<<(128 * 128 + 255) / 256, 256, 0, stream>>>(W2, wb2, 100, 100, 128, 128, 0.1f);
  k_cvt<<<(128 * 1216 + 255) / 256, 256, 0, stream>>>(W3, wb3, 100, 1216, 128, 1216, 0.1f);
  k_edge_prep<<<(N_EDGES + 255) / 256, 256, 0, stream>>>(dd, rv, basis, Yb);
  k_zero<<<(N_NODES * 72 + 255) / 256, 256, 0, stream>>>(out, N_NODES * 72);
  k_gemm_silu<<<N_EDGES / 16, 256, 0, stream>>>(basis, wb0, h1, 32);
  k_gemm_silu<<<N_EDGES / 16, 256, 0, stream>>>(h1, wb1, h2, 128);
  k_gemm_silu<<<N_EDGES / 16, 256, 0, stream>>>(h2, wb2, h3, 128);
  k_tp<<<N_EDGES / 16, 128, 0, stream>>>(h3, wb3, cg, Yb, x, ei, out);
}
